// MultiHeadAttention_71210557768281
// MI455X (gfx1250) — matrix operand order repaired and checked by reading, not compiled, not run on hardware
//
#include <hip/hip_runtime.h>
#include <hip/hip_bf16.h>

// ---------------- problem constants ----------------
#define HEADS  3
#define DH     64
#define NB     8
#define NSEQ   1024
#define DIMC   192
#define INNERC 192
#define QK_SCALE 0.125f   // 64^-0.5

typedef __attribute__((ext_vector_type(16))) _Float16 v16h;
typedef __attribute__((ext_vector_type(8)))  float    v8f;
typedef __attribute__((ext_vector_type(4)))  _Float16 h4;
typedef __attribute__((ext_vector_type(4)))  float    f4;

// ================= WMMA fragment loaders (wave32, V_WMMA_F32_16X16X32_F16) =================
// A (16x32 MxK): lanes 0-15 -> M=lane, K=0..15; lanes 16-31 -> M=lane-16, K=16..31
// Two contiguous 16B reads per lane -> ds_load_b128 pairs.
__device__ __forceinline__ v16h frag_a_rm(const _Float16* t, int row0, int k0, int ld) {
  const int lane = threadIdx.x & 31;
  const _Float16* p = t + (size_t)(row0 + (lane & 15)) * ld + k0 + ((lane >> 4) << 4);
  v16h f;
#pragma unroll
  for (int e = 0; e < 8; ++e) f[e] = p[e];
#pragma unroll
  for (int e = 0; e < 8; ++e) f[8 + e] = p[8 + e];
  return f;
}
// B (32x16 KxN): lanes 0-15 -> N=lane, K=0..15; lanes 16-31 -> K=16..31.
// Expects B staged TRANSPOSED (n-major: t[n*ld + k]) so each lane reads 16 contiguous f16
// (two ds_load_b128) instead of 16 strided ds_load_u16 + v_mov_b16 packing.
__device__ __forceinline__ v16h frag_b_nm(const _Float16* t, int col0, int k0, int ld) {
  const int lane = threadIdx.x & 31;
  const _Float16* p = t + (size_t)(col0 + (lane & 15)) * ld + k0 + ((lane >> 4) << 4);
  v16h f;
#pragma unroll
  for (int e = 0; e < 16; ++e) f[e] = p[e];
  return f;
}
#define WMMA_F16(a, b, c) \
  __builtin_amdgcn_wmma_f32_16x16x32_f16(false, (a), false, (b), (short)0, (c), false, false)

// ================= 1) QKV projection: x(8192x192) @ W(192x576) -> f16 q/k/v (B,H,N,Dh) ==========
__global__ void __launch_bounds__(256) qkv_gemm(const float* __restrict__ x,
                                                const float* __restrict__ Wq,
                                                _Float16* __restrict__ qh,
                                                _Float16* __restrict__ kh,
                                                _Float16* __restrict__ vh) {
  __shared__ alignas(16) _Float16 As[128 * 32];
  __shared__ alignas(16) _Float16 Bs[64 * 32];   // n-major (transposed) B tile
  const int tid = threadIdx.x, wv = tid >> 5, lane = tid & 31;
  const int m0 = blockIdx.x * 128;
  const int c0 = blockIdx.y * 64;
  v8f acc[4] = {};
  for (int k0 = 0; k0 < DIMC; k0 += 32) {
    // stage A tile 128x32 (f32 -> f16), row-major
    for (int i = tid; i < 128 * 8; i += 256) {
      int r = i >> 3, c = i & 7;
      f4 v = ((const f4*)(x + (size_t)(m0 + r) * DIMC + k0))[c];
      h4 hv; hv.x = (_Float16)v.x; hv.y = (_Float16)v.y; hv.z = (_Float16)v.z; hv.w = (_Float16)v.w;
      ((h4*)As)[i] = hv;
    }
    // stage B tile 32x64 (f32 -> f16), transposed to n-major Bs[n][k]
    for (int i = tid; i < 32 * 16; i += 256) {
      int k = i >> 4, c = i & 15;
      f4 v = ((const f4*)(Wq + (size_t)(k0 + k) * 576 + c0))[c];
      int n = c << 2;
      Bs[(n + 0) * 32 + k] = (_Float16)v.x;
      Bs[(n + 1) * 32 + k] = (_Float16)v.y;
      Bs[(n + 2) * 32 + k] = (_Float16)v.z;
      Bs[(n + 3) * 32 + k] = (_Float16)v.w;
    }
    __syncthreads();
    v16h a = frag_a_rm(As, wv * 16, 0, 32);
#pragma unroll
    for (int nt = 0; nt < 4; ++nt) {
      v16h b = frag_b_nm(Bs, nt * 16, 0, 32);
      acc[nt] = WMMA_F16(a, b, acc[nt]);
    }
    __syncthreads();
  }
  const int rbase = (lane >> 4) << 3, nn = lane & 15;
#pragma unroll
  for (int nt = 0; nt < 4; ++nt)
#pragma unroll
    for (int r = 0; r < 8; ++r) {
      int m = m0 + wv * 16 + rbase + r;
      int c = c0 + nt * 16 + nn;
      float val = acc[nt][r];
      int part = c / INNERC, wi = c % INNERC;
      int h = wi >> 6, dh = wi & 63;
      int b = m >> 10, i = m & 1023;
      _Float16* dst = (part == 0) ? qh : (part == 1) ? kh : vh;
      if (part == 0) val *= QK_SCALE;
      dst[(((size_t)(b * HEADS + h) << 10) + i) * DH + dh] = (_Float16)val;
    }
}

// ================= 2) scores = q @ k^T (pre-scaled), per (b,h): 1024x1024x64 ====================
__global__ void __launch_bounds__(256) scores_gemm(const _Float16* __restrict__ qh,
                                                   const _Float16* __restrict__ kh,
                                                   float* __restrict__ scores) {
  __shared__ alignas(16) _Float16 Qs[128 * 64];
  __shared__ alignas(16) _Float16 Ks[64 * 64];   // k rows == B^T rows -> already fragment-friendly
  const int tid = threadIdx.x, wv = tid >> 5, lane = tid & 31;
  const int n0 = blockIdx.x * 64;
  const int m0 = blockIdx.y * 128;
  const int bh = blockIdx.z;
  const _Float16* q = qh + ((size_t)bh << 10) * DH;
  const _Float16* k = kh + ((size_t)bh << 10) * DH;
  const h4* qg = (const h4*)(q + (size_t)m0 * DH);
  const h4* kg = (const h4*)(k + (size_t)n0 * DH);
  for (int i = tid; i < 128 * 16; i += 256) ((h4*)Qs)[i] = qg[i];
  for (int i = tid; i < 64 * 16;  i += 256) ((h4*)Ks)[i] = kg[i];
  __syncthreads();
  v8f acc[4] = {};
#pragma unroll
  for (int kk = 0; kk < DH; kk += 32) {
    v16h a = frag_a_rm(Qs, wv * 16, kk, DH);
#pragma unroll
    for (int nt = 0; nt < 4; ++nt) {
      v16h b = frag_b_nm(Ks, nt * 16, kk, DH);
      acc[nt] = WMMA_F16(a, b, acc[nt]);
    }
  }
  const int rbase = (lane >> 4) << 3, nn = lane & 15;
  float* out = scores + ((size_t)bh << 20);
#pragma unroll
  for (int nt = 0; nt < 4; ++nt)
#pragma unroll
    for (int r = 0; r < 8; ++r)
      out[((size_t)(m0 + wv * 16 + rbase + r) << 10) + n0 + nt * 16 + nn] = acc[nt][r];
}

// ================= 3) depthwise 3x3 conv + BN + ReLU on (B*H, 1024, 1024) =======================
__global__ void dwconv_bn_relu(const float* __restrict__ in, float* __restrict__ out,
                               const float* __restrict__ w, const float* __restrict__ cb,
                               const float* __restrict__ g, const float* __restrict__ bb) {
  const unsigned idx = blockIdx.x * blockDim.x + threadIdx.x; // B*H*N*N = 25,165,824
  const int j  = idx & 1023;
  const int i  = (idx >> 10) & 1023;
  const int bh = idx >> 20;
  const int h  = bh % HEADS;
  const float* img = in + ((size_t)bh << 20);
  float acc = 0.f;
#pragma unroll
  for (int di = -1; di <= 1; ++di) {
    int ii = i + di;
    if (ii < 0 || ii > 1023) continue;
#pragma unroll
    for (int dj = -1; dj <= 1; ++dj) {
      int jj = j + dj;
      if (jj < 0 || jj > 1023) continue;
      acc += img[((size_t)ii << 10) + jj] * w[h * 9 + (di + 1) * 3 + (dj + 1)];
    }
  }
  acc += cb[h];
  float s = g[h] * rsqrtf(1.0f + 1e-5f);
  float y = acc * s + bb[h];
  out[idx] = y > 0.f ? y : 0.f;
}

// ================= 4a) global mean pool per (b,h) ===============================================
__global__ void pool_mean(const float* __restrict__ a, float* __restrict__ pooled) {
  const int bh = blockIdx.x;
  const f4* p = (const f4*)(a + ((size_t)bh << 20));
  float s = 0.f;
  for (int i = threadIdx.x; i < (1 << 18); i += 256) {
    f4 v = p[i];
    s += v.x + v.y + v.z + v.w;
  }
  __shared__ float red[256];
  red[threadIdx.x] = s;
  __syncthreads();
  for (int st = 128; st > 0; st >>= 1) {
    if ((int)threadIdx.x < st) red[threadIdx.x] += red[threadIdx.x + st];
    __syncthreads();
  }
  if (threadIdx.x == 0) pooled[bh] = red[0] * (1.0f / 1048576.0f);
}

// ================= 4b) SE gate: 3 -> 1 -> 3, sigmoid ============================================
__global__ void se_gate(const float* __restrict__ pooled,
                        const float* __restrict__ w1, const float* __restrict__ b1,
                        const float* __restrict__ w2, const float* __restrict__ b2,
                        float* __restrict__ cw) {
  int t = threadIdx.x;
  if (t >= NB * HEADS) return;
  int b = t / HEADS, h = t % HEADS;
  float hid = b1[0];
#pragma unroll
  for (int j = 0; j < HEADS; ++j) hid += pooled[b * HEADS + j] * w1[j];
  hid = fmaxf(hid, 0.f);
  float z = hid * w2[h] + b2[h];
  cw[t] = 1.0f / (1.0f + __expf(-z));
}

// ================= 5) fused cw-scale + row softmax -> f16 attn ==================================
__global__ void __launch_bounds__(256) softmax_rows(const float* __restrict__ a,
                                                    const float* __restrict__ cw,
                                                    _Float16* __restrict__ attn) {
  const int row = blockIdx.x;          // bh*1024 + i
  const float c = cw[row >> 10];
  f4 v = ((const f4*)(a + ((size_t)row << 10)))[threadIdx.x];
  v.x *= c; v.y *= c; v.z *= c; v.w *= c;
  float m = fmaxf(fmaxf(v.x, v.y), fmaxf(v.z, v.w));
  __shared__ float red[256];
  red[threadIdx.x] = m;
  __syncthreads();
  for (int st = 128; st > 0; st >>= 1) {
    if ((int)threadIdx.x < st) red[threadIdx.x] = fmaxf(red[threadIdx.x], red[threadIdx.x + st]);
    __syncthreads();
  }
  m = red[0];
  __syncthreads();
  f4 e;
  e.x = __expf(v.x - m); e.y = __expf(v.y - m); e.z = __expf(v.z - m); e.w = __expf(v.w - m);
  red[threadIdx.x] = e.x + e.y + e.z + e.w;
  __syncthreads();
  for (int st = 128; st > 0; st >>= 1) {
    if ((int)threadIdx.x < st) red[threadIdx.x] += red[threadIdx.x + st];
    __syncthreads();
  }
  float inv = 1.0f / red[0];
  h4 o;
  o.x = (_Float16)(e.x * inv); o.y = (_Float16)(e.y * inv);
  o.z = (_Float16)(e.z * inv); o.w = (_Float16)(e.w * inv);
  ((h4*)(attn + ((size_t)row << 10)))[threadIdx.x] = o;
}

// ================= 6) out = attn @ v, per (b,h): 1024 x 64 x 1024 ===============================
__global__ void __launch_bounds__(256) attnv_gemm(const _Float16* __restrict__ attn,
                                                  const _Float16* __restrict__ vh,
                                                  _Float16* __restrict__ outh) {
  __shared__ alignas(16) _Float16 As[128 * 32];
  __shared__ alignas(16) _Float16 Vs[64 * 32];   // n-major (transposed) v tile
  const int tid = threadIdx.x, wv = tid >> 5, lane = tid & 31;
  const int m0 = blockIdx.x * 128;
  const int bh = blockIdx.y;
  const _Float16* A = attn + ((size_t)bh << 20);
  const _Float16* V = vh + ((size_t)bh << 10) * DH;
  v8f acc[4] = {};
  for (int k0 = 0; k0 < NSEQ; k0 += 32) {
    for (int i = tid; i < 128 * 8; i += 256) {   // attn tile 128x32 (h4 units), row-major
      int r = i >> 3, c = i & 7;
      ((h4*)As)[i] = ((const h4*)(A + (size_t)(m0 + r) * NSEQ + k0))[c];
    }
    for (int i = tid; i < 32 * 16; i += 256) {   // v tile 32x64 -> transposed Vs[n][k]
      int k = i >> 4, c = i & 15;
      h4 hv = ((const h4*)(V + (size_t)(k0 + k) * DH))[c];
      int n = c << 2;
      Vs[(n + 0) * 32 + k] = hv.x;
      Vs[(n + 1) * 32 + k] = hv.y;
      Vs[(n + 2) * 32 + k] = hv.z;
      Vs[(n + 3) * 32 + k] = hv.w;
    }
    __syncthreads();
    v16h a = frag_a_rm(As, wv * 16, 0, 32);
#pragma unroll
    for (int nt = 0; nt < 4; ++nt) {
      v16h b = frag_b_nm(Vs, nt * 16, 0, 32);
      acc[nt] = WMMA_F16(a, b, acc[nt]);
    }
    __syncthreads();
  }
  const int rbase = (lane >> 4) << 3, nn = lane & 15;
  const int b = bh / HEADS, h = bh % HEADS;
#pragma unroll
  for (int nt = 0; nt < 4; ++nt)
#pragma unroll
    for (int r = 0; r < 8; ++r) {
      int i  = m0 + wv * 16 + rbase + r;
      int dh = nt * 16 + nn;
      outh[((size_t)(b << 10) + i) * INNERC + h * DH + dh] = (_Float16)acc[nt][r];
    }
}

// ================= 7) final projection: (8192x192) @ (192x192) + bias -> f32 ====================
__global__ void __launch_bounds__(256) proj_gemm(const _Float16* __restrict__ outh,
                                                 const float* __restrict__ Wout,
                                                 const float* __restrict__ bout,
                                                 float* __restrict__ y) {
  __shared__ alignas(16) _Float16 As[128 * 32];
  __shared__ alignas(16) _Float16 Bs[64 * 32];   // n-major (transposed) W tile
  const int tid = threadIdx.x, wv = tid >> 5, lane = tid & 31;
  const int m0 = blockIdx.x * 128;
  const int c0 = blockIdx.y * 64;
  v8f acc[4] = {};
  for (int k0 = 0; k0 < INNERC; k0 += 32) {
    for (int i = tid; i < 128 * 8; i += 256) {
      int r = i >> 3, c = i & 7;
      ((h4*)As)[i] = ((const h4*)(outh + (size_t)(m0 + r) * INNERC + k0))[c];
    }
    for (int i = tid; i < 32 * 16; i += 256) {
      int k = i >> 4, c = i & 15;
      f4 v = ((const f4*)(Wout + (size_t)(k0 + k) * DIMC + c0))[c];
      int n = c << 2;
      Bs[(n + 0) * 32 + k] = (_Float16)v.x;
      Bs[(n + 1) * 32 + k] = (_Float16)v.y;
      Bs[(n + 2) * 32 + k] = (_Float16)v.z;
      Bs[(n + 3) * 32 + k] = (_Float16)v.w;
    }
    __syncthreads();
    v16h a = frag_a_rm(As, wv * 16, 0, 32);
#pragma unroll
    for (int nt = 0; nt < 4; ++nt) {
      v16h b = frag_b_nm(Bs, nt * 16, 0, 32);
      acc[nt] = WMMA_F16(a, b, acc[nt]);
    }
    __syncthreads();
  }
  const int rbase = (lane >> 4) << 3, nn = lane & 15;
#pragma unroll
  for (int nt = 0; nt < 4; ++nt)
#pragma unroll
    for (int r = 0; r < 8; ++r) {
      int m = m0 + wv * 16 + rbase + r;
      int c = c0 + nt * 16 + nn;
      y[(size_t)m * DIMC + c] = acc[nt][r] + bout[c];
    }
}

// ================= host side ====================================================================
extern "C" void kernel_launch(void* const* d_in, const int* in_sizes, int n_in,
                              void* d_out, int out_size, void* d_ws, size_t ws_size,
                              hipStream_t stream) {
  const float* x       = (const float*)d_in[0];
  const float* W_qkv   = (const float*)d_in[1];
  const float* conv1_w = (const float*)d_in[2];
  const float* conv1_b = (const float*)d_in[3];
  const float* bn1_g   = (const float*)d_in[4];
  const float* bn1_b   = (const float*)d_in[5];
  const float* conv2_w = (const float*)d_in[6];
  const float* conv2_b = (const float*)d_in[7];
  const float* bn2_g   = (const float*)d_in[8];
  const float* bn2_b   = (const float*)d_in[9];
  const float* ca_w1   = (const float*)d_in[10];
  const float* ca_b1   = (const float*)d_in[11];
  const float* ca_w2   = (const float*)d_in[12];
  const float* ca_b2   = (const float*)d_in[13];
  const float* W_out   = (const float*)d_in[14];
  const float* b_out   = (const float*)d_in[15];
  float* out = (float*)d_out;

  // workspace layout (bytes)
  char* ws = (char*)d_ws;
  const size_t QKV_BYTES = (size_t)NB * HEADS * NSEQ * DH * 2;  // 3,145,728 (f16)
  const size_t QH_OFF  = 0;
  const size_t KH_OFF  = QH_OFF + QKV_BYTES;
  const size_t VH_OFF  = KH_OFF + QKV_BYTES;
  const size_t OH_OFF  = VH_OFF + QKV_BYTES;                    // f16 (B,N,192)
  const size_t PL_OFF  = OH_OFF + QKV_BYTES;
  const size_t CW_OFF  = PL_OFF + 256;
  const size_t S_OFF   = CW_OFF + 256;                          // fp32 (B,H,N,N) 100.7 MB
  const size_t MAP_BYTES = (size_t)NB * HEADS * NSEQ * NSEQ * 4;
  const size_t A1_OFF  = S_OFF + MAP_BYTES;                     // fp32 scratch map / f16 attn alias

  _Float16* qh    = (_Float16*)(ws + QH_OFF);
  _Float16* kh    = (_Float16*)(ws + KH_OFF);
  _Float16* vh    = (_Float16*)(ws + VH_OFF);
  _Float16* outh  = (_Float16*)(ws + OH_OFF);
  float*    pooled= (float*)   (ws + PL_OFF);
  float*    cw    = (float*)   (ws + CW_OFF);
  float*    smap  = (float*)   (ws + S_OFF);    // scores, then conv2 output (a2)
  float*    amap  = (float*)   (ws + A1_OFF);   // conv1 output (a1)
  _Float16* attn  = (_Float16*)(ws + A1_OFF);   // f16 attn aliases dead a1

  const int BH = NB * HEADS;                    // 24
  const unsigned MAP_ELEMS = (unsigned)BH << 20;

  // 1) qkv projection (folds q scale), f16 q/k/v
  qkv_gemm<<<dim3((NB * NSEQ) / 128, 576 / 64), 256, 0, stream>>>(x, W_qkv, qh, kh, vh);
  // 2) scores = q @ k^T
  scores_gemm<<<dim3(NSEQ / 64, NSEQ / 128, BH), 256, 0, stream>>>(qh, kh, smap);
  // 3) two depthwise conv + BN + ReLU stages (ping-pong smap <-> amap)
  dwconv_bn_relu<<<MAP_ELEMS / 256, 256, 0, stream>>>(smap, amap, conv1_w, conv1_b, bn1_g, bn1_b);
  dwconv_bn_relu<<<MAP_ELEMS / 256, 256, 0, stream>>>(amap, smap, conv2_w, conv2_b, bn2_g, bn2_b);
  // 4) SE gate
  pool_mean<<<BH, 256, 0, stream>>>(smap, pooled);
  se_gate<<<1, 32, 0, stream>>>(pooled, ca_w1, ca_b1, ca_w2, ca_b2, cw);
  // 5) fused scale + softmax -> f16 attn
  softmax_rows<<<BH * NSEQ, 256, 0, stream>>>(smap, cw, attn);
  // 6) attn @ v -> f16 (B,N,192)
  attnv_gemm<<<dim3(NSEQ / 128, BH), 256, 0, stream>>>(attn, vh, outh);
  // 7) final projection + bias -> fp32 output
  proj_gemm<<<dim3((NB * NSEQ) / 128, DIMC / 64), 256, 0, stream>>>(outh, W_out, b_out, out);
}
